// DgCnnPN_60146722013628
// MI455X (gfx1250) — matrix operand order repaired and checked by reading, not compiled, not run on hardware
//
#include <hip/hip_runtime.h>
#include <hip/hip_bf16.h>
#include <math.h>

// ---------------------------------------------------------------------------
// Problem constants (from reference)
// ---------------------------------------------------------------------------
static constexpr int B_   = 64;
static constexpr int S_   = 512;
static constexpr int D_   = 256;
static constexpr int CE_  = 256;
static constexpr int PE_  = 512;          // CE + WE
static constexpr int P2_  = 100;          // 2*P
static constexpr int F_   = 768;          // 3*D
static constexpr int HALO_= 8;            // max dilation offset is 5; pad to 8
static constexpr int SP_  = S_ + 2*HALO_; // padded sequence length = 528
static constexpr float EPS_ = 1e-3f;

// conv blocking
static constexpr int MB_   = 32;          // positions per block
static constexpr int SROWS_= 44;          // staged rows: [s_base-6, s_base+37]
static constexpr int CINMAX_ = 512;

// ---------------------------------------------------------------------------
// WMMA types (CDNA5 gfx1250, wave32)
// ---------------------------------------------------------------------------
typedef __attribute__((ext_vector_type(16))) __bf16        v16bf;
typedef __attribute__((ext_vector_type(8)))  float         v8f;
typedef __attribute__((ext_vector_type(4)))  unsigned int  u32x4;

union Frag { u32x4 u[2]; v16bf v; };

__device__ __forceinline__ unsigned short f2bf(float f) {
  unsigned int u = __builtin_bit_cast(unsigned int, f);
  unsigned int r = u + 0x7FFFu + ((u >> 16) & 1u);   // round-to-nearest-even
  return (unsigned short)(r >> 16);
}

__device__ __forceinline__ float sigmoidf_(float x) {
  return 1.0f / (1.0f + __expf(-x));
}

// ---------------------------------------------------------------------------
// Zero fill (for padded bf16 buffers: halos must be 0 for SAME padding)
// ---------------------------------------------------------------------------
__global__ void zero_kernel(unsigned int* p, long long ndw) {
  long long i = (long long)blockIdx.x * blockDim.x + threadIdx.x;
  long long st = (long long)gridDim.x * blockDim.x;
  for (; i < ndw; i += st) p[i] = 0u;
}

// ---------------------------------------------------------------------------
// Pack a K x N f32 matrix (row stride N) into bf16 WMMA B-fragments.
// Fragment (kt, nt), lane l, halfword h:
//   K = kt*32 + (l>>4)*16 + h ; N = nt*16 + (l&15)
// contiguous k order (not AMD's table order); the A loads use the same order. Out-of-range -> 0.
// ---------------------------------------------------------------------------
__global__ void pack_frags_kernel(const float* __restrict__ W,
                                  unsigned short* __restrict__ out,
                                  int K, int N, int KT, int NT) {
  int total = KT * NT * 32;
  for (int idx = blockIdx.x * blockDim.x + threadIdx.x; idx < total;
       idx += gridDim.x * blockDim.x) {
    int l  = idx & 31;
    int f  = idx >> 5;
    int nt = f % NT;
    int kt = f / NT;
    int n  = nt * 16 + (l & 15);
    int kb = kt * 32 + (l >> 4) * 16;
    unsigned short* dst = out + (size_t)f * 512 + l * 16;
    #pragma unroll
    for (int h = 0; h < 16; ++h) {
      int k = kb + h;
      float v = (k < K && n < N) ? W[(size_t)k * N + n] : 0.0f;
      dst[h] = f2bf(v);
    }
  }
}

// ---------------------------------------------------------------------------
// Embedding: embed[b,s,:] = concat(char_table[tok], word_table[wtok]) + pos_table[pos]
// stored as bf16 into padded (B, SP, PE) buffer (interior only; halo stays 0).
// ---------------------------------------------------------------------------
__global__ void embed_kernel(const int* __restrict__ inputs,
                             const int* __restrict__ inputs_word,
                             const int* __restrict__ inputs_pos,
                             const float* __restrict__ char_table,
                             const float* __restrict__ word_table,
                             const float* __restrict__ pos_table,
                             unsigned short* __restrict__ embed_bf) {
  long long total = (long long)B_ * S_ * PE_;
  long long st = (long long)gridDim.x * blockDim.x;
  for (long long i = (long long)blockIdx.x * blockDim.x + threadIdx.x; i < total; i += st) {
    int c = (int)(i % PE_);
    long long bs = i / PE_;
    int s = (int)(bs % S_);
    int b = (int)(bs / S_);
    int tok  = inputs[bs];
    int wtok = inputs_word[bs];
    int pos  = inputs_pos[bs];
    float v = (c < CE_) ? char_table[(size_t)tok * CE_ + c]
                        : word_table[(size_t)wtok * CE_ + (c - CE_)];
    v += pos_table[(size_t)pos * PE_ + c];
    embed_bf[((size_t)b * SP_ + s + HALO_) * PE_ + c] = f2bf(v);
  }
}

// ---------------------------------------------------------------------------
// mask = (inputs != 0); written to workspace (f32) and to d_out (f32 1.0/0.0)
// ---------------------------------------------------------------------------
__global__ void mask_kernel(const int* __restrict__ inputs,
                            float* __restrict__ mask_ws,
                            float* __restrict__ mask_out) {
  int i = blockIdx.x * blockDim.x + threadIdx.x;
  if (i < B_ * S_) {
    float m = (inputs[i] != 0) ? 1.0f : 0.0f;
    mask_ws[i]  = m;
    mask_out[i] = m;
  }
}

// ---------------------------------------------------------------------------
// Fused dilated gated conv layer (WMMA bf16, f32 accumulate), LDS-staged.
//
// Block (256 thr = 8 waves) owns 32 consecutive positions x all 512 outputs.
// Step 1: cooperatively stage the 44-row x Cin bf16 input slab into LDS
//         (covers taps -5..+5 for all dilations; contiguous in padded buffer).
// Step 2: wave w computes M=32 x channel-pair tiles ct=2w, 2w+1
//         (gate cols ct*16.., value cols 256+ct*16..): 8 accumulators.
//         Per K-step: 8 WMMAs, 4 ds_read_b128 (A), 8 global b128 (B).
// Step 3: fused sigmoid gate + residual + mask; store f32 master + padded bf16.
// Grid must be exactly B*S/32 = 1024 blocks.
// ---------------------------------------------------------------------------
__global__ __launch_bounds__(256)
void dgc_conv_kernel(const unsigned short* __restrict__ A,   // padded bf16 (B,SP,Cin)
                     const float* __restrict__ xin,          // (B,S,D) residual or null
                     const unsigned short* __restrict__ Wf,  // packed frags
                     const float* __restrict__ bias,         // (512,)
                     const float* __restrict__ mask,         // (B,S)
                     float* __restrict__ xout,               // (B,S,D)
                     unsigned short* __restrict__ xout_bf,   // padded bf16 (B,SP,D)
                     int Cin, int dil, int residual) {
  __shared__ unsigned short smem[SROWS_ * CINMAX_];   // 45056 B max

  const int KT = Cin >> 5;
  int b      = blockIdx.x >> 4;          // 16 position tiles of 32 per batch row
  int s_base = (blockIdx.x & 15) << 5;

  // ---- stage input slab: padded rows [s_base+2, s_base+45] (i.e. s-6..s+37) ----
  {
    const u32x4* src = reinterpret_cast<const u32x4*>(
        A + ((size_t)b * SP_ + s_base + 2) * Cin);
    u32x4* dst = reinterpret_cast<u32x4*>(smem);
    int nchunk = SROWS_ * Cin / 8;       // 16B chunks
    for (int i = threadIdx.x; i < nchunk; i += 256) dst[i] = src[i];
  }
  __syncthreads();

  int wave = threadIdx.x >> 5;
  int lane = threadIdx.x & 31;
  int row  = lane & 15;
  int half = lane >> 4;
  int ct0  = wave * 2;                   // two channel-pair tiles per wave

  v8f acc[2][4];                         // [mm][g0,h0,g1,h1]
  #pragma unroll
  for (int mm = 0; mm < 2; ++mm)
    #pragma unroll
    for (int j = 0; j < 4; ++j)
      acc[mm][j] = v8f{0.f,0.f,0.f,0.f,0.f,0.f,0.f,0.f};

  for (int tap = 0; tap < 3; ++tap) {
    int srow0 = row + (tap - 1) * dil + 6;     // LDS row for mm=0 (range 1..42)
    const unsigned short* wtap = Wf + (size_t)tap * KT * 32 * 512;
    for (int kt = 0; kt < KT; ++kt) {
      int k0 = kt << 5;
      const unsigned short* wk = wtap + (size_t)kt * 32 * 512;

      // prefetch next K-step's weight fragments (L2-resident stream)
      if (kt + 1 < KT) {
        __builtin_prefetch(wk + (size_t)32 * 512 + (size_t)ct0 * 512 + lane * 16, 0, 1);
      }

      Frag fa[2];
      #pragma unroll
      for (int mm = 0; mm < 2; ++mm) {
        const unsigned short* ar = smem + (size_t)(srow0 + 16 * mm) * Cin + k0;
        fa[mm].u[0] = *reinterpret_cast<const u32x4*>(ar + half * 16);
        fa[mm].u[1] = *reinterpret_cast<const u32x4*>(ar + half * 16 + 8);
      }

      Frag fb[4];
      #pragma unroll
      for (int j = 0; j < 4; ++j) {
        int ctj = ct0 + (j >> 1) + ((j & 1) ? 16 : 0);   // {ct0, ct0+16, ct0+1, ct0+17}
        const u32x4* bp = reinterpret_cast<const u32x4*>(
            wk + (size_t)ctj * 512 + lane * 16);
        fb[j].u[0] = bp[0];
        fb[j].u[1] = bp[1];
      }

      #pragma unroll
      for (int mm = 0; mm < 2; ++mm) {
        acc[mm][0] = __builtin_amdgcn_wmma_f32_16x16x32_bf16(
            false, fa[mm].v, false, fb[0].v, (short)0, acc[mm][0], false, false); // g ct0
        acc[mm][1] = __builtin_amdgcn_wmma_f32_16x16x32_bf16(
            false, fa[mm].v, false, fb[1].v, (short)0, acc[mm][1], false, false); // h ct0
        acc[mm][2] = __builtin_amdgcn_wmma_f32_16x16x32_bf16(
            false, fa[mm].v, false, fb[2].v, (short)0, acc[mm][2], false, false); // g ct1
        acc[mm][3] = __builtin_amdgcn_wmma_f32_16x16x32_bf16(
            false, fa[mm].v, false, fb[3].v, (short)0, acc[mm][3], false, false); // h ct1
      }
    }
  }

  // ---- epilogue: gate + residual + mask; C layout: elem i -> M = i + half*8 ----
  int n = lane & 15;
  #pragma unroll
  for (int ci = 0; ci < 2; ++ci) {
    int c = (ct0 + ci) * 16 + n;
    float bgs = bias[c];
    float bhs = bias[256 + c];
    #pragma unroll
    for (int mm = 0; mm < 2; ++mm) {
      v8f ag = acc[mm][ci * 2 + 0];
      v8f ah = acc[mm][ci * 2 + 1];
      int mbase = 16 * mm + half * 8;
      #pragma unroll
      for (int i = 0; i < 8; ++i) {
        int pos = s_base + mbase + i;
        float g  = sigmoidf_(ag[i] + bgs);
        float hv = ah[i] + bhs;
        float m  = mask[b * S_ + pos];
        float out;
        if (residual) {
          float xr = xin[((size_t)b * S_ + pos) * D_ + c];
          out = (xr * (1.0f - g) + hv * g) * m;
        } else {
          out = hv * g * m;
        }
        xout[((size_t)b * S_ + pos) * D_ + c] = out;
        xout_bf[((size_t)b * SP_ + pos + HALO_) * D_ + c] = f2bf(out);
      }
    }
  }
}

// ---------------------------------------------------------------------------
// sub head: sigmoid(x @ sub_w + sub_b), written transposed (B,2,S)
// ---------------------------------------------------------------------------
__global__ void sub_head_kernel(const float* __restrict__ x,
                                const float* __restrict__ sub_w,
                                const float* __restrict__ sub_b,
                                float* __restrict__ out) {
  int i = blockIdx.x * blockDim.x + threadIdx.x;
  if (i >= B_ * S_) return;
  int b = i / S_, s = i % S_;
  const float* xr = x + (size_t)i * D_;
  float d0 = 0.f, d1 = 0.f;
  for (int c = 0; c < D_; ++c) {
    float v = xr[c];
    d0 += v * sub_w[c * 2 + 0];
    d1 += v * sub_w[c * 2 + 1];
  }
  out[((size_t)b * 2 + 0) * S_ + s] = sigmoidf_(d0 + sub_b[0]);
  out[((size_t)b * 2 + 1) * S_ + s] = sigmoidf_(d1 + sub_b[1]);
}

// ---------------------------------------------------------------------------
// vec[b, j] = x[b, loc0, j] (j<256)  |  x[b, loc1, j-256]
// ---------------------------------------------------------------------------
__global__ void vec_kernel(const float* __restrict__ x,
                           const int* __restrict__ loc,
                           float* __restrict__ vec) {
  int i = blockIdx.x * blockDim.x + threadIdx.x;
  if (i >= B_ * 2 * D_) return;
  int b = i >> 9, j = i & 511;
  int s = (j < D_) ? loc[b * 2 + 0] : loc[b * 2 + 1];
  int c = (j < D_) ? j : j - D_;
  vec[i] = x[((size_t)b * S_ + s) * D_ + c];
}

// ---------------------------------------------------------------------------
// BN stats over feat = concat(x, broadcast vec). For f<256: mean/var over B*S
// of x; for f>=256 the value is constant over S, so stats reduce over B only.
// ---------------------------------------------------------------------------
__global__ void bn_stats_kernel(const float* __restrict__ x,
                                const float* __restrict__ vec,
                                float* __restrict__ mu,
                                float* __restrict__ var) {
  __shared__ float ssum[256];
  __shared__ float ssq[256];
  int f = blockIdx.x;
  float sum = 0.f, sq = 0.f;
  if (f < D_) {
    for (int i = threadIdx.x; i < B_ * S_; i += blockDim.x) {
      float v = x[(size_t)i * D_ + f];
      sum += v; sq += v * v;
    }
  } else {
    int j = f - D_;
    for (int b = threadIdx.x; b < B_; b += blockDim.x) {
      float v = vec[b * 512 + j];
      sum += v; sq += v * v;
    }
  }
  ssum[threadIdx.x] = sum; ssq[threadIdx.x] = sq;
  __syncthreads();
  for (int off = 128; off > 0; off >>= 1) {
    if ((int)threadIdx.x < off) {
      ssum[threadIdx.x] += ssum[threadIdx.x + off];
      ssq[threadIdx.x]  += ssq[threadIdx.x + off];
    }
    __syncthreads();
  }
  if (threadIdx.x == 0) {
    float cnt = (f < D_) ? (float)(B_ * S_) : (float)B_;
    float m = ssum[0] / cnt;
    mu[f]  = m;
    var[f] = ssq[0] / cnt - m * m;
  }
}

// ---------------------------------------------------------------------------
// feat_norm[b,s,f] = ((feat - mu) * rsqrt(var+eps) * gamma + beta) -> bf16
// ---------------------------------------------------------------------------
__global__ void feat_norm_kernel(const float* __restrict__ x,
                                 const float* __restrict__ vec,
                                 const float* __restrict__ mu,
                                 const float* __restrict__ var,
                                 const float* __restrict__ gamma,
                                 const float* __restrict__ beta,
                                 unsigned short* __restrict__ feat) {
  long long total = (long long)B_ * S_ * F_;
  long long st = (long long)gridDim.x * blockDim.x;
  for (long long i = (long long)blockIdx.x * blockDim.x + threadIdx.x; i < total; i += st) {
    int f = (int)(i % F_);
    long long bs = i / F_;
    int b = (int)(bs / S_);
    float v = (f < D_) ? x[bs * D_ + f] : vec[b * 512 + (f - D_)];
    float fn = (v - mu[f]) * rsqrtf(var[f] + EPS_) * gamma[f] + beta[f];
    feat[i] = f2bf(fn);
  }
}

// ---------------------------------------------------------------------------
// po head GEMM via WMMA: (B*S, 768) @ (768, 112pad) -> sigmoid -> transposed
// (B, 100, S) write. Predication only AFTER the WMMA loop.
// grid supplies exactly 2048*7 waves (1792 blocks x 8 waves).
// ---------------------------------------------------------------------------
__global__ __launch_bounds__(256)
void po_head_kernel(const unsigned short* __restrict__ feat,
                    const unsigned short* __restrict__ Wf,
                    const float* __restrict__ po_b,
                    float* __restrict__ out) {
  const int KT = 24, NT = 7;
  int gw   = blockIdx.x * 8 + (threadIdx.x >> 5);
  int lane = threadIdx.x & 31;
  int nt = gw % NT;
  int mt = gw / NT;
  int b      = mt >> 5;
  int s_base = (mt & 31) << 4;
  int row  = lane & 15;
  int half = lane >> 4;
  const unsigned short* arow = feat + ((size_t)b * S_ + s_base + row) * F_;

  v8f acc = {0.f,0.f,0.f,0.f,0.f,0.f,0.f,0.f};
  for (int kt = 0; kt < KT; ++kt) {
    int k0 = kt << 5;
    Frag fa;
    fa.u[0] = *reinterpret_cast<const u32x4*>(arow + k0 + half * 16);
    fa.u[1] = *reinterpret_cast<const u32x4*>(arow + k0 + half * 16 + 8);
    const u32x4* bp = reinterpret_cast<const u32x4*>(
        Wf + ((size_t)kt * NT + nt) * 512 + lane * 16);
    Frag fb; fb.u[0] = bp[0]; fb.u[1] = bp[1];
    acc = __builtin_amdgcn_wmma_f32_16x16x32_bf16(
        false, fa.v, false, fb.v, (short)0, acc, false, false);
  }

  int n = lane & 15;
  int p = nt * 16 + n;
  if (p < P2_) {
    float bb = po_b[p];
    int m0 = half * 8;
    #pragma unroll
    for (int i = 0; i < 8; ++i) {
      int s = s_base + m0 + i;
      out[((size_t)b * P2_ + p) * S_ + s] = sigmoidf_(acc[i] + bb);
    }
  }
}

// ---------------------------------------------------------------------------
// Host-side launch
// ---------------------------------------------------------------------------
extern "C" void kernel_launch(void* const* d_in, const int* in_sizes, int n_in,
                              void* d_out, int out_size, void* d_ws, size_t ws_size,
                              hipStream_t stream) {
  (void)in_sizes; (void)n_in; (void)out_size; (void)ws_size;

  const int*   inputs        = (const int*)  d_in[0];
  const int*   inputs_word   = (const int*)  d_in[1];
  const int*   inputs_pos    = (const int*)  d_in[2];
  const int*   input_sub_loc = (const int*)  d_in[3];
  const float* char_table    = (const float*)d_in[4];
  const float* word_table    = (const float*)d_in[5];
  const float* pos_table     = (const float*)d_in[6];
  const float* conv_w1       = (const float*)d_in[7];
  const float* conv_b1       = (const float*)d_in[8];
  const float* conv_w        = (const float*)d_in[9];
  const float* conv_b        = (const float*)d_in[10];
  const float* sub_w         = (const float*)d_in[11];
  const float* sub_b         = (const float*)d_in[12];
  const float* po_w          = (const float*)d_in[13];
  const float* po_b          = (const float*)d_in[14];
  const float* bn_gamma      = (const float*)d_in[15];
  const float* bn_beta       = (const float*)d_in[16];

  float* out_sub  = (float*)d_out;                          // (B,2,S)
  float* out_po   = out_sub + (size_t)B_ * 2 * S_;          // (B,100,S)
  float* out_mask = out_po  + (size_t)B_ * P2_ * S_;        // (B,S) as 1.0/0.0

  // ---- workspace layout (256B-aligned slices) ----
  char* ws = (char*)d_ws;
  size_t off = 0;
  auto take = [&](size_t bytes) -> char* {
    char* p = ws + off;
    off += (bytes + 255) & ~(size_t)255;
    return p;
  };
  float*          mask_ws  = (float*)         take((size_t)B_ * S_ * 4);
  unsigned short* embed_bf = (unsigned short*)take((size_t)B_ * SP_ * PE_ * 2);
  unsigned short* xb_a     = (unsigned short*)take((size_t)B_ * SP_ * D_ * 2);
  unsigned short* xb_b     = (unsigned short*)take((size_t)B_ * SP_ * D_ * 2);
  float*          xf_a     = (float*)         take((size_t)B_ * S_ * D_ * 4);
  float*          xf_b     = (float*)         take((size_t)B_ * S_ * D_ * 4);
  unsigned short* w1f      = (unsigned short*)take((size_t)3 * 16 * 32 * 1024);
  unsigned short* wf       = (unsigned short*)take((size_t)11 * 3 * 8 * 32 * 1024);
  unsigned short* pof      = (unsigned short*)take((size_t)24 * 7 * 1024);
  float*          vec      = (float*)         take((size_t)B_ * 512 * 4);
  float*          mu       = (float*)         take((size_t)F_ * 4);
  float*          var      = (float*)         take((size_t)F_ * 4);
  unsigned short* feat     = (unsigned short*)take((size_t)B_ * S_ * F_ * 2);

  // ---- zero padded bf16 activation buffers (halos must be 0) ----
  zero_kernel<<<2048, 256, 0, stream>>>((unsigned int*)embed_bf,
                                        (long long)B_ * SP_ * PE_ / 2);
  zero_kernel<<<1024, 256, 0, stream>>>((unsigned int*)xb_a,
                                        (long long)B_ * SP_ * D_ / 2);
  zero_kernel<<<1024, 256, 0, stream>>>((unsigned int*)xb_b,
                                        (long long)B_ * SP_ * D_ / 2);

  // ---- pack weights into WMMA B-fragments (bf16) ----
  for (int t = 0; t < 3; ++t) {
    pack_frags_kernel<<<64, 256, 0, stream>>>(
        conv_w1 + (size_t)t * PE_ * 512,
        w1f + (size_t)t * 16 * 32 * 512, 512, 512, 16, 32);
  }
  for (int li = 0; li < 11; ++li) {
    for (int t = 0; t < 3; ++t) {
      pack_frags_kernel<<<32, 256, 0, stream>>>(
          conv_w + ((size_t)(li * 3 + t)) * D_ * 512,
          wf + (size_t)(li * 3 + t) * 8 * 32 * 512, 256, 512, 8, 32);
    }
  }
  pack_frags_kernel<<<21, 256, 0, stream>>>(po_w, pof, 768, 100, 24, 7);

  // ---- embedding + mask ----
  embed_kernel<<<4096, 256, 0, stream>>>(inputs, inputs_word, inputs_pos,
                                         char_table, word_table, pos_table,
                                         embed_bf);
  mask_kernel<<<128, 256, 0, stream>>>(inputs, mask_ws, out_mask);

  // ---- 12 gated conv layers (grid = B*S/32 = 1024 blocks) ----
  static const int DILS[12] = {1, 2, 5, 1, 2, 5, 1, 2, 5, 1, 1, 1};

  // layer 1: Cin = 512 (embed), no residual
  dgc_conv_kernel<<<1024, 256, 0, stream>>>(embed_bf, nullptr, w1f, conv_b1,
                                            mask_ws, xf_a, xb_a,
                                            PE_, DILS[0], 0);

  unsigned short* a_bf = xb_a;  unsigned short* o_bf = xb_b;
  float*          a_f  = xf_a;  float*          o_f  = xf_b;
  for (int i = 0; i < 11; ++i) {
    dgc_conv_kernel<<<1024, 256, 0, stream>>>(
        a_bf, a_f, wf + (size_t)i * 3 * 8 * 32 * 512, conv_b + i * 512,
        mask_ws, o_f, o_bf, D_, DILS[i + 1], 1);
    unsigned short* tb = a_bf; a_bf = o_bf; o_bf = tb;
    float*          tf = a_f;  a_f  = o_f;  o_f  = tf;
  }
  const float* x_final = a_f;   // final activation after swap

  // ---- heads + BN ----
  sub_head_kernel<<<128, 256, 0, stream>>>(x_final, sub_w, sub_b, out_sub);
  vec_kernel<<<128, 256, 0, stream>>>(x_final, input_sub_loc, vec);
  bn_stats_kernel<<<768, 256, 0, stream>>>(x_final, vec, mu, var);
  feat_norm_kernel<<<4096, 256, 0, stream>>>(x_final, vec, mu, var,
                                             bn_gamma, bn_beta, feat);
  po_head_kernel<<<1792, 256, 0, stream>>>(feat, pof, po_b, out_po);
}
